// DvneGCN_73143293051581
// MI455X (gfx1250) — matrix operand order repaired and checked by reading, not compiled, not run on hardware
//
#include <hip/hip_runtime.h>

// ---------------------------------------------------------------------------
// DVNE-GCN forward for MI455X (gfx1250, wave32).
//   GEMMs : v_wmma_f32_16x16x32_bf16, 2-term bf16 split (~fp32 accuracy),
//           compile-time K/NOUT, b128 A loads, pre-packed B fragments.
//   SpMM  : per-(edge, float4-chunk) gather + global_atomic_add_f32 scatter.
// ---------------------------------------------------------------------------

#define F_DIM 128
#define H_DIM 32
#define C_DIM 16

typedef __attribute__((ext_vector_type(16))) __bf16 v16bf;
typedef __attribute__((ext_vector_type(8)))  float  v8f;

__device__ __forceinline__ unsigned short f2bf_bits(float f) {
  union { float f; unsigned u; } x; x.f = f;
  unsigned r = x.u + 0x7fffu + ((x.u >> 16) & 1u);   // round-to-nearest-even
  return (unsigned short)(r >> 16);
}
__device__ __forceinline__ float bf2f(unsigned short s) {
  union { unsigned u; float f; } y; y.u = ((unsigned)s) << 16; return y.f;
}
__device__ __forceinline__ __bf16 bits2bf(unsigned short s) {
  union { unsigned short s; __bf16 b; } y; y.s = s; return y.b;
}

// Pack W[K,NOUT] (row-major fp32) into B-fragment order, split bf16 hi/lo.
// Element index: ((tn*KSTEPS + ks)*32 + lane)*16 + e ;  K of element = ks*32 + 8*g + (e&7) + 16*(e>>3).
__global__ void pack_weights(const float* __restrict__ W, unsigned short* __restrict__ Bhi,
                             unsigned short* __restrict__ Blo, int K, int Nout) {
  const int ksteps = (K + 31) / 32;
  const int total = (Nout / 16) * ksteps * 32 * 16;
  const int i = blockIdx.x * blockDim.x + threadIdx.x;
  if (i >= total) return;
  const int e    = i & 15;
  const int lane = (i >> 4) & 31;
  const int frag = i >> 9;
  const int ks = frag % ksteps;
  const int tn = frag / ksteps;
  const int g  = lane >> 4;
  const int lm = lane & 15;
  const int kb = ks * 32 + 8 * g + (e & 7) + ((e >> 3) << 4);
  const float f = (kb < K) ? W[(size_t)kb * Nout + tn * 16 + lm] : 0.0f;
  const unsigned short h = f2bf_bits(f);
  Bhi[i] = h;
  Blo[i] = f2bf_bits(f - bf2f(h));
}

// C[M,NOUT] = A[M,K] @ W ; one wave per 16x16 tile; K,NOUT compile-time.
template <int K, int NOUT>
__global__ __launch_bounds__(256)
void gemm_wmma_packed(const float* __restrict__ A,
                      const unsigned short* __restrict__ Bhi,
                      const unsigned short* __restrict__ Blo,
                      float* __restrict__ Cout, int M) {
  constexpr int KSTEPS = (K + 31) / 32;
  constexpr int TILES_N = NOUT / 16;
  const int wavesPerBlock = blockDim.x >> 5;
  const int waveId = blockIdx.x * wavesPerBlock + (threadIdx.x >> 5);
  const int tilesM = M >> 4;                     // M = 50000 = 3125 * 16 exactly
  if (waveId >= tilesM * TILES_N) return;        // wave-uniform: EXEC stays all-1s
  const int tm = waveId / TILES_N;
  const int tn = waveId % TILES_N;
  const int lane = threadIdx.x & 31;
  const int g  = lane >> 4;
  const int lm = lane & 15;

  v8f c = {};
  const float* Arow = A + (size_t)(tm * 16 + lm) * K;

#pragma unroll
  for (int ks = 0; ks < KSTEPS; ++ks) {
    const int k0 = ks * 32;
    // A fragment: per lane, two contiguous 8-float runs (ISA 16-bit A layout:
    // e 0..7 -> K = k0+8g+e ; e 8..15 -> K = k0+16+8g+(e-8)).
    float af[16];
    {
      const float4* ap = (const float4*)(Arow + k0 + 8 * g);   // 16B aligned
      const float4 r0 = ap[0], r1 = ap[1];
      af[0] = r0.x; af[1] = r0.y; af[2] = r0.z; af[3] = r0.w;
      af[4] = r1.x; af[5] = r1.y; af[6] = r1.z; af[7] = r1.w;
    }
    if (k0 + 16 < K) {                                          // folds per K
      const float4* ap = (const float4*)(Arow + k0 + 16 + 8 * g);
      const float4 r0 = ap[0], r1 = ap[1];
      af[8]  = r0.x; af[9]  = r0.y; af[10] = r0.z; af[11] = r0.w;
      af[12] = r1.x; af[13] = r1.y; af[14] = r1.z; af[15] = r1.w;
    } else {
#pragma unroll
      for (int j = 8; j < 16; ++j) af[j] = 0.0f;
    }

    v16bf a_hi, a_lo;
#pragma unroll
    for (int e = 0; e < 16; ++e) {
      const unsigned short h = f2bf_bits(af[e]);
      a_hi[e] = bits2bf(h);
      a_lo[e] = bits2bf(f2bf_bits(af[e] - bf2f(h)));
    }

    // Pre-packed B fragment: 32 bytes contiguous per lane.
    const size_t fbase = ((size_t)(tn * KSTEPS + ks) * 32 + lane) * 16;
    const v16bf b_hi = *(const v16bf*)(Bhi + fbase);
    const v16bf b_lo = *(const v16bf*)(Blo + fbase);

    c = __builtin_amdgcn_wmma_f32_16x16x32_bf16(false, a_hi, false, b_hi, (short)0, c, false, false);
    c = __builtin_amdgcn_wmma_f32_16x16x32_bf16(false, a_hi, false, b_lo, (short)0, c, false, false);
    c = __builtin_amdgcn_wmma_f32_16x16x32_bf16(false, a_lo, false, b_hi, (short)0, c, false, false);
  }

#pragma unroll
  for (int r = 0; r < 8; ++r) {                 // D layout: VGPR r -> M = 8*g + r
    const int m = tm * 16 + 8 * g + r;
    Cout[(size_t)m * NOUT + tn * 16 + lm] = c[r];
  }
}

// out[dst] += w * sup[src] over edges; one thread per (edge, 4-float chunk).
template <int D>
__global__ void spmm_scatter(const int* __restrict__ src, const int* __restrict__ dst,
                             const float* __restrict__ w, const float* __restrict__ sup,
                             float* out, int E) {
  constexpr int CH = D / 4;
  const long long t = (long long)blockIdx.x * blockDim.x + threadIdx.x;
  const long long total = (long long)E * CH;
  if (t >= total) return;
  const int e  = (int)(t / CH);
  const int ch = (int)(t % CH);
  const int s = src[e], d = dst[e];
  const float wt = w[e];
  const float4 v = *reinterpret_cast<const float4*>(sup + (size_t)s * D + ch * 4);
  float* op = out + (size_t)d * D + ch * 4;
  atomicAdd(op + 0, wt * v.x);
  atomicAdd(op + 1, wt * v.y);
  atomicAdd(op + 2, wt * v.z);
  atomicAdd(op + 3, wt * v.w);
}

__global__ void zero_f32(float* p, long long n) {
  long long i = (long long)blockIdx.x * blockDim.x + threadIdx.x;
  if (i < n) p[i] = 0.0f;
}

__global__ void relu_inplace(float* p, long long n) {
  long long i = (long long)blockIdx.x * blockDim.x + threadIdx.x;
  if (i < n) p[i] = fmaxf(p[i], 0.0f);
}

// std = elu(sraw)+1 ; enc = means + std * rand_u
__global__ void encode_kernel(const float* __restrict__ means, const float* __restrict__ sraw,
                              const float* __restrict__ randu,
                              float* __restrict__ std_out, float* __restrict__ enc,
                              long long n) {
  long long i = (long long)blockIdx.x * blockDim.x + threadIdx.x;
  if (i < n) {
    const float s = sraw[i];
    const float st = (s > 0.0f) ? (s + 1.0f) : expf(s);
    std_out[i] = st;
    enc[i] = means[i] + st * randu[i];
  }
}

static inline int packed_elems(int K, int Nout) {
  return (Nout / 16) * ((K + 31) / 32) * 32 * 16;
}

extern "C" void kernel_launch(void* const* d_in, const int* in_sizes, int n_in,
                              void* d_out, int out_size, void* d_ws, size_t ws_size,
                              hipStream_t stream) {
  const float* x    = (const float*)d_in[0];
  const int*   esrc = (const int*)  d_in[1];
  const int*   edst = (const int*)  d_in[2];
  const float* ew   = (const float*)d_in[3];
  const float* ru   = (const float*)d_in[4];
  const float* W1   = (const float*)d_in[5];
  const float* Wm   = (const float*)d_in[6];
  const float* Wsm  = (const float*)d_in[7];
  const float* Wd   = (const float*)d_in[8];
  const float* Wo   = (const float*)d_in[9];
  const int E = in_sizes[1];
  const int N = in_sizes[4] / C_DIM;

  // d_out = prediction [N*128] | means [N*16] | std [N*16]
  float* pred  = (float*)d_out;
  float* means = pred  + (size_t)N * F_DIM;
  float* stdo  = means + (size_t)N * C_DIM;

  // workspace: 256*N floats (~51 MB) + ~40 KB packed weights
  float* ws     = (float*)d_ws;
  float* bufG   = ws;                              // N*128: GEMM outputs (reused)
  float* hidden = bufG   + (size_t)N * F_DIM;      // N*32
  float* dec    = hidden + (size_t)N * H_DIM;      // N*32
  float* bufGm  = dec    + (size_t)N * H_DIM;      // N*16
  float* bufGs  = bufGm  + (size_t)N * C_DIM;      // N*16
  float* sraw   = bufGs  + (size_t)N * C_DIM;      // N*16
  float* enc    = sraw   + (size_t)N * C_DIM;      // N*16

  unsigned short* pw = (unsigned short*)(enc + (size_t)N * C_DIM);
  const int p1 = packed_elems(F_DIM, H_DIM);   // W1: 4096
  const int pm = packed_elems(H_DIM, C_DIM);   // Wm: 512
  const int pd = packed_elems(C_DIM, H_DIM);   // Wd: 1024
  const int po = packed_elems(H_DIM, F_DIM);   // Wo: 4096
  unsigned short* W1hi = pw;            unsigned short* W1lo = W1hi + p1;
  unsigned short* Wmhi = W1lo + p1;     unsigned short* Wmlo = Wmhi + pm;
  unsigned short* Wshi = Wmlo + pm;     unsigned short* Wslo = Wshi + pm;
  unsigned short* Wdhi = Wslo + pm;     unsigned short* Wdlo = Wdhi + pd;
  unsigned short* Wohi = Wdlo + pd;     unsigned short* Wolo = Wohi + po;

  auto pack = [&](const float* W, unsigned short* hi, unsigned short* lo, int K, int Nout) {
    const int total = packed_elems(K, Nout);
    pack_weights<<<(total + 255) / 256, 256, 0, stream>>>(W, hi, lo, K, Nout);
  };
  auto zero = [&](float* p, long long n) {
    zero_f32<<<(int)((n + 255) / 256), 256, 0, stream>>>(p, n);
  };
  auto relu = [&](float* p, long long n) {
    relu_inplace<<<(int)((n + 255) / 256), 256, 0, stream>>>(p, n);
  };

  const int tilesM = N / 16;

  // Pack all weights into B-fragment bf16 hi/lo form.
  pack(W1,  W1hi, W1lo, F_DIM, H_DIM);
  pack(Wm,  Wmhi, Wmlo, H_DIM, C_DIM);
  pack(Wsm, Wshi, Wslo, H_DIM, C_DIM);
  pack(Wd,  Wdhi, Wdlo, C_DIM, H_DIM);
  pack(Wo,  Wohi, Wolo, H_DIM, F_DIM);

  // Layer 1: hidden = relu(spmm(x @ W1))
  gemm_wmma_packed<F_DIM, H_DIM><<<(tilesM * (H_DIM / 16) + 7) / 8, 256, 0, stream>>>(
      x, W1hi, W1lo, bufG, N);
  zero(hidden, (long long)N * H_DIM);
  spmm_scatter<H_DIM><<<(int)(((long long)E * (H_DIM / 4) + 255) / 256), 256, 0, stream>>>(
      esrc, edst, ew, bufG, hidden, E);
  relu(hidden, (long long)N * H_DIM);

  // means = spmm(hidden @ Wm) ; std = elu(spmm(hidden @ Ws)) + 1
  gemm_wmma_packed<H_DIM, C_DIM><<<(tilesM * (C_DIM / 16) + 7) / 8, 256, 0, stream>>>(
      hidden, Wmhi, Wmlo, bufGm, N);
  gemm_wmma_packed<H_DIM, C_DIM><<<(tilesM * (C_DIM / 16) + 7) / 8, 256, 0, stream>>>(
      hidden, Wshi, Wslo, bufGs, N);
  zero(means, (long long)N * C_DIM);
  zero(sraw,  (long long)N * C_DIM);
  spmm_scatter<C_DIM><<<(int)(((long long)E * (C_DIM / 4) + 255) / 256), 256, 0, stream>>>(
      esrc, edst, ew, bufGm, means, E);
  spmm_scatter<C_DIM><<<(int)(((long long)E * (C_DIM / 4) + 255) / 256), 256, 0, stream>>>(
      esrc, edst, ew, bufGs, sraw, E);
  encode_kernel<<<(int)(((long long)N * C_DIM + 255) / 256), 256, 0, stream>>>(
      means, sraw, ru, stdo, enc, (long long)N * C_DIM);

  // decoded = relu(spmm(enc @ Wd))
  gemm_wmma_packed<C_DIM, H_DIM><<<(tilesM * (H_DIM / 16) + 7) / 8, 256, 0, stream>>>(
      enc, Wdhi, Wdlo, bufG, N);
  zero(dec, (long long)N * H_DIM);
  spmm_scatter<H_DIM><<<(int)(((long long)E * (H_DIM / 4) + 255) / 256), 256, 0, stream>>>(
      esrc, edst, ew, bufG, dec, E);
  relu(dec, (long long)N * H_DIM);

  // prediction = relu(spmm(decoded @ Wo))
  gemm_wmma_packed<H_DIM, F_DIM><<<(tilesM * (F_DIM / 16) + 7) / 8, 256, 0, stream>>>(
      dec, Wohi, Wolo, bufG, N);
  zero(pred, (long long)N * F_DIM);
  spmm_scatter<F_DIM><<<(int)(((long long)E * (F_DIM / 4) + 255) / 256), 256, 0, stream>>>(
      esrc, edst, ew, bufG, pred, E);
  relu(pred, (long long)N * F_DIM);
}
